// TransformerBlock_31310311588425
// MI455X (gfx1250) — compile-verified, matrix operand order checked by reading, not run on hardware
//
#include <hip/hip_runtime.h>

// ---------------------------------------------------------------------------
// Transformer block for MI455X (gfx1250): bf16 WMMA everywhere, fp32 residual.
// GEMM uses the Tensor Data Mover (TDM) to stream tiles into LDS
// (double-buffered, TENSORcnt-tracked) while waves do only ds_load + wmma.
// B=2, T=2048, DIM=2048, NH=16, HD=128, HIDDEN=8192.
// ---------------------------------------------------------------------------

#define DIMC   2048
#define NH     16
#define HD     128
#define HIDDENC 8192
#define TT     2048
#define BT     4096   // B*T rows

typedef __attribute__((ext_vector_type(16))) __bf16 bf16x16;
typedef __attribute__((ext_vector_type(8)))  float  f32x8;
typedef unsigned int u32x4 __attribute__((ext_vector_type(4)));
typedef int          i32x4 __attribute__((ext_vector_type(4)));
typedef int          i32x8 __attribute__((ext_vector_type(8)));

union AF { bf16x16 v; unsigned int u[8]; };
union CF { f32x8  v; float        f[8]; };

__device__ __forceinline__ unsigned short f2bf(float f) {
    union { float f; unsigned u; } x; x.f = f;
    unsigned r = x.u + 0x7FFFu + ((x.u >> 16) & 1u);   // round-to-nearest-even
    return (unsigned short)(r >> 16);
}
__device__ __forceinline__ float bf2f(unsigned short b) {
    union { unsigned u; float f; } x; x.u = ((unsigned)b) << 16; return x.f;
}

// ---------------------------------------------------------------------------
// TDM: issue a 2D tile load (rows x cols of bf16) global -> LDS.
//   lds_addr : LDS byte address of tile start (row-contiguous destination)
//   gptr     : global address of tile start (within the tensor)
//   td0/td1  : tensor dims (elements per row = K, number of rows)
//   tl0/tl1  : tile dims   (cols = 32, rows = 128)
//   stride0  : tensor row stride in elements
// D# bit packing per CDNA5 ISA ch.8 (group0 128b, group1 256b).
// This toolchain's builtin takes 6 args (g0, g1, g2, g3, extra, cpol);
// groups 2/3 + extra are zero (unused for a 2D tile).
// ---------------------------------------------------------------------------
__device__ __forceinline__ void tdm_load_tile(
    unsigned lds_addr, const unsigned short* gptr,
    unsigned td0, unsigned td1, unsigned tl0, unsigned tl1, unsigned stride0)
{
    unsigned long long ga = (unsigned long long)(size_t)gptr;
    u32x4 g0;
    g0.x = 1u;                                      // count=1, user descriptor
    g0.y = lds_addr;                                // lds_addr [63:32]
    g0.z = (unsigned)ga;                            // global_addr [95:64]
    g0.w = (unsigned)(ga >> 32) | (2u << 30);       // global_addr hi | type=2
    i32x8 g1;
    g1[0] = (int)(1u << 16);                        // data_size=1 (2 bytes)
    g1[1] = (int)((td0 & 0xFFFFu) << 16);           // tensor_dim0[15:0] @48
    g1[2] = (int)(((td0 >> 16) & 0xFFFFu) | ((td1 & 0xFFFFu) << 16));
    g1[3] = (int)(((td1 >> 16) & 0xFFFFu) | (tl0 << 16));   // tile_dim0 @112
    g1[4] = (int)tl1;                               // tile_dim1 @128 (dim2=0)
    g1[5] = (int)stride0;                           // tensor_dim0_stride[31:0]
    g1[6] = 0;
    g1[7] = 0;
    i32x4 z4 = {0, 0, 0, 0};                        // groups 2/3 unused (2D)
    i32x8 z8 = {0, 0, 0, 0, 0, 0, 0, 0};
    __builtin_amdgcn_tensor_load_to_lds(g0, g1, z4, z4, z8, 0);
}

// ---------------------------------------------------------------------------
// RMSNorm (fp32 in) -> bf16 out.  One block per row of 2048.
// ---------------------------------------------------------------------------
__global__ void __launch_bounds__(256) rmsnorm_cast(
    const float* __restrict__ x, const float* __restrict__ w,
    unsigned short* __restrict__ out)
{
    const int row = blockIdx.x;
    const float* xr = x + (size_t)row * DIMC;
    __shared__ float red[8];
    float s = 0.f;
    for (int c = threadIdx.x; c < DIMC; c += 256) { float v = xr[c]; s += v * v; }
    #pragma unroll
    for (int m = 16; m >= 1; m >>= 1) s += __shfl_xor(s, m, 32);
    if ((threadIdx.x & 31) == 0) red[threadIdx.x >> 5] = s;
    __syncthreads();
    float tot = 0.f;
    #pragma unroll
    for (int i = 0; i < 8; ++i) tot += red[i];
    const float rinv = rsqrtf(tot / (float)DIMC + 1e-5f);
    unsigned short* orow = out + (size_t)row * DIMC;
    for (int c = threadIdx.x; c < DIMC; c += 256)
        orow[c] = f2bf(xr[c] * rinv * w[c]);
}

// ---------------------------------------------------------------------------
// W [K][N] fp32  ->  WT [N][K] bf16  (so GEMM B-fragments gather like A).
// grid = (N/32, K/32), 256 threads.
// ---------------------------------------------------------------------------
__global__ void __launch_bounds__(256) cast_transpose(
    const float* __restrict__ W, unsigned short* __restrict__ WT,
    int K, int N)
{
    __shared__ float tile[32][33];
    const int k0 = blockIdx.y * 32, n0 = blockIdx.x * 32;
    const int tx = threadIdx.x & 31, ty = threadIdx.x >> 5;  // 8 rows/pass
    #pragma unroll
    for (int i = 0; i < 4; ++i) {
        int kr = ty + i * 8;
        tile[kr][tx] = W[(size_t)(k0 + kr) * N + n0 + tx];
    }
    __syncthreads();
    #pragma unroll
    for (int i = 0; i < 4; ++i) {
        int nr = ty + i * 8;
        WT[(size_t)(n0 + nr) * K + k0 + tx] = f2bf(tile[tx][nr]);
    }
}

// ---------------------------------------------------------------------------
// bf16 GEMM:  C[M][N] = A[M][K] * Bt[N][K]^T   (Bt row = output column).
// Block tile 128x128, 8 waves (4 along M x 2 along N), each wave 32x64
// = 2x4 WMMA accumulators.  K consumed in 32-deep slices, double-buffered
// in LDS and streamed by the Tensor Data Mover; wave 0 issues the DMA and
// gates the block on s_wait_tensorcnt.
// ---------------------------------------------------------------------------
__global__ void __launch_bounds__(256) gemm_bf16(
    const unsigned short* __restrict__ A,
    const unsigned short* __restrict__ Bt,
    unsigned short* __restrict__ C,
    int M, int N, int K)
{
    __shared__ unsigned short As[2][128][32];
    __shared__ unsigned short Bs[2][128][32];
    const int tid  = threadIdx.x;
    const int wv   = tid >> 5, lane = tid & 31;
    const int hl   = lane >> 4, l16 = lane & 15;
    const int m0   = blockIdx.y * 128, n0 = blockIdx.x * 128;
    const int wm   = wv & 3, wn = wv >> 2;

    const unsigned ldsA[2] = { (unsigned)(size_t)&As[0][0][0],
                               (unsigned)(size_t)&As[1][0][0] };
    const unsigned ldsB[2] = { (unsigned)(size_t)&Bs[0][0][0],
                               (unsigned)(size_t)&Bs[1][0][0] };
    const unsigned short* Ab = A  + (size_t)m0 * K;
    const unsigned short* Bb = Bt + (size_t)n0 * K;
    const int nk = K >> 5;

    CF acc[2][4];
    #pragma unroll
    for (int i = 0; i < 2; ++i)
        #pragma unroll
        for (int j = 0; j < 4; ++j)
            #pragma unroll
            for (int r = 0; r < 8; ++r) acc[i][j].f[r] = 0.f;

    // prologue: stream slices 0 and 1 into the two LDS buffers
    if (wv == 0) {
        tdm_load_tile(ldsA[0], Ab,      (unsigned)K, (unsigned)M, 32, 128, (unsigned)K);
        tdm_load_tile(ldsB[0], Bb,      (unsigned)K, (unsigned)N, 32, 128, (unsigned)K);
        if (nk > 1) {
            tdm_load_tile(ldsA[1], Ab + 32, (unsigned)K, (unsigned)M, 32, 128, (unsigned)K);
            tdm_load_tile(ldsB[1], Bb + 32, (unsigned)K, (unsigned)N, 32, 128, (unsigned)K);
        }
    }

    for (int i = 0; i < nk; ++i) {
        const int buf = i & 1;
        if (wv == 0) {
            if (i + 1 < nk) __builtin_amdgcn_s_wait_tensorcnt(2);  // slice i done
            else            __builtin_amdgcn_s_wait_tensorcnt(0);
        }
        __syncthreads();

        #pragma unroll
        for (int mt = 0; mt < 2; ++mt) {
            AF a;
            const int arow = wm * 32 + mt * 16 + l16;
            #pragma unroll
            for (int v = 0; v < 8; ++v) {
                int kk = (v >> 2) * 16 + hl * 8 + (v & 3) * 2;
                a.u[v] = *(const unsigned int*)&As[buf][arow][kk];
            }
            #pragma unroll
            for (int nt = 0; nt < 4; ++nt) {
                AF b;
                const int brow = wn * 64 + nt * 16 + l16;
                #pragma unroll
                for (int v = 0; v < 8; ++v) {
                    int kk = (v >> 2) * 16 + hl * 8 + (v & 3) * 2;
                    b.u[v] = *(const unsigned int*)&Bs[buf][brow][kk];
                }
                acc[mt][nt].v = __builtin_amdgcn_wmma_f32_16x16x32_bf16(
                    false, a.v, false, b.v, (short)0, acc[mt][nt].v, false, false);
            }
        }
        __syncthreads();                                   // block done reading buf
        if (wv == 0 && i + 2 < nk) {                       // refill buf with slice i+2
            tdm_load_tile(ldsA[buf], Ab + (size_t)(i + 2) * 32,
                          (unsigned)K, (unsigned)M, 32, 128, (unsigned)K);
            tdm_load_tile(ldsB[buf], Bb + (size_t)(i + 2) * 32,
                          (unsigned)K, (unsigned)N, 32, 128, (unsigned)K);
        }
    }

    #pragma unroll
    for (int mt = 0; mt < 2; ++mt)
        #pragma unroll
        for (int nt = 0; nt < 4; ++nt)
            #pragma unroll
            for (int r = 0; r < 8; ++r) {
                int row = m0 + wm * 32 + mt * 16 + r + hl * 8;
                int col = n0 + wn * 64 + nt * 16 + l16;
                C[(size_t)row * N + col] = f2bf(acc[mt][nt].f[r]);
            }
}

// ---------------------------------------------------------------------------
// RoPE in-place on bf16 [B*T][DIM]; one thread per (b,t,h,i<64) pair.
// ---------------------------------------------------------------------------
__global__ void rope_inplace(unsigned short* __restrict__ q,
                             const float* __restrict__ cs,
                             const float* __restrict__ sn)
{
    int e = blockIdx.x * blockDim.x + threadIdx.x;     // 2^22 total
    int i = e & 63;
    int h = (e >> 6) & 15;
    int t = (e >> 10) & (TT - 1);
    int b = e >> 21;
    size_t row = ((size_t)b * TT + t) * DIMC + h * HD;
    float x1 = bf2f(q[row + i]), x2 = bf2f(q[row + 64 + i]);
    float c = cs[t * 64 + i], s = sn[t * 64 + i];
    q[row + i]      = f2bf(x1 * c - x2 * s);
    q[row + 64 + i] = f2bf(x1 * s + x2 * c);
}

// v [B][T][H*HD] -> vT [B][H][HD][T]
__global__ void transpose_v(const unsigned short* __restrict__ v,
                            unsigned short* __restrict__ vT)
{
    int e = blockIdx.x * blockDim.x + threadIdx.x;     // 2^23 total
    int d = e & (HD - 1);
    int h = (e >> 7) & 15;
    int t = (e >> 11) & (TT - 1);
    int b = e >> 22;
    vT[(((size_t)(b * NH + h)) * HD + d) * TT + t] = v[e];
}

// ---------------------------------------------------------------------------
// Flash attention: 4 waves per block, each wave owns 16 query rows;
// online softmax over 32-key blocks; S and PV both via bf16 WMMA.
// grid = (T/64, B*NH), 128 threads.
// ---------------------------------------------------------------------------
__global__ void __launch_bounds__(128) attention(
    const unsigned short* __restrict__ q,
    const unsigned short* __restrict__ k,
    const unsigned short* __restrict__ vT,
    unsigned short* __restrict__ y)
{
    __shared__ unsigned short Pst[4][16][32];
    const int tid  = threadIdx.x;
    const int wv   = tid >> 5, lane = tid & 31;
    const int hl   = lane >> 4, l16 = lane & 15;
    const int bh   = blockIdx.y, b = bh >> 4, h = bh & 15;
    const int qbase = blockIdx.x * 64 + wv * 16;
    const float scale = 0.08838834764831845f;          // 1/sqrt(128)

    const unsigned short* qp = q  + ((size_t)b * TT) * DIMC + h * HD;
    const unsigned short* kp = k  + ((size_t)b * TT) * DIMC + h * HD;
    const unsigned short* vp = vT + ((size_t)bh) * HD * TT;

    AF qf[4];                                          // Q rows, D in 4 chunks
    {
        const int arow = qbase + l16;
        #pragma unroll
        for (int c = 0; c < 4; ++c)
            #pragma unroll
            for (int v = 0; v < 8; ++v) {
                int kk = (v >> 2) * 16 + hl * 8 + (v & 3) * 2;
                qf[c].u[v] = *(const unsigned int*)&qp[(size_t)arow * DIMC + c * 32 + kk];
            }
    }

    CF O[8];
    #pragma unroll
    for (int nt = 0; nt < 8; ++nt)
        #pragma unroll
        for (int r = 0; r < 8; ++r) O[nt].f[r] = 0.f;
    float mrow[8], lrow[8];
    #pragma unroll
    for (int r = 0; r < 8; ++r) { mrow[r] = -3.0e38f; lrow[r] = 0.f; }

    const int jmax = (qbase + 15) >> 5;
    for (int j = 0; j <= jmax; ++j) {
        const int jbase = j << 5;
        CF S[2];
        #pragma unroll
        for (int t = 0; t < 2; ++t)
            #pragma unroll
            for (int r = 0; r < 8; ++r) S[t].f[r] = 0.f;

        #pragma unroll
        for (int t = 0; t < 2; ++t) {
            const int krow = jbase + t * 16 + l16;
            #pragma unroll
            for (int c = 0; c < 4; ++c) {
                AF bf;
                #pragma unroll
                for (int v = 0; v < 8; ++v) {
                    int kk = (v >> 2) * 16 + hl * 8 + (v & 3) * 2;
                    bf.u[v] = *(const unsigned int*)&kp[(size_t)krow * DIMC + c * 32 + kk];
                }
                S[t].v = __builtin_amdgcn_wmma_f32_16x16x32_bf16(
                    false, qf[c].v, false, bf.v, (short)0, S[t].v, false, false);
            }
        }
        // scale + causal mask
        const bool need_mask = (jbase + 31) > qbase;
        #pragma unroll
        for (int t = 0; t < 2; ++t)
            #pragma unroll
            for (int r = 0; r < 8; ++r) {
                float sv = S[t].f[r] * scale;
                if (need_mask) {
                    int rowg = qbase + r + hl * 8;
                    int colg = jbase + t * 16 + l16;
                    if (colg > rowg) sv = -3.0e38f;
                }
                S[t].f[r] = sv;
            }
        // online softmax: row max across the 16-lane half + both tiles
        float alpha[8];
        #pragma unroll
        for (int r = 0; r < 8; ++r) {
            float mx = fmaxf(S[0].f[r], S[1].f[r]);
            #pragma unroll
            for (int md = 8; md >= 1; md >>= 1)
                mx = fmaxf(mx, __shfl_xor(mx, md, 32));
            float mnew = fmaxf(mrow[r], mx);
            alpha[r] = __expf(mrow[r] - mnew);
            mrow[r]  = mnew;
        }
        // P = exp(S - m), row sums, stage P into LDS (C-layout -> A-layout)
        #pragma unroll
        for (int r = 0; r < 8; ++r) {
            float p0 = __expf(S[0].f[r] - mrow[r]);
            float p1 = __expf(S[1].f[r] - mrow[r]);
            float ps = p0 + p1;
            #pragma unroll
            for (int md = 8; md >= 1; md >>= 1)
                ps += __shfl_xor(ps, md, 32);
            lrow[r] = lrow[r] * alpha[r] + ps;
            const int prow = r + hl * 8;
            Pst[wv][prow][l16]      = f2bf(p0);
            Pst[wv][prow][16 + l16] = f2bf(p1);
        }
        // rescale O by alpha
        #pragma unroll
        for (int nt = 0; nt < 8; ++nt)
            #pragma unroll
            for (int r = 0; r < 8; ++r) O[nt].f[r] *= alpha[r];
        // P as A-fragment (wave-local LDS, DS ops are in-order within a wave)
        AF pf;
        #pragma unroll
        for (int v = 0; v < 8; ++v) {
            int kk = (v >> 2) * 16 + hl * 8 + (v & 3) * 2;
            pf.u[v] = *(const unsigned int*)&Pst[wv][l16][kk];
        }
        // O += P * V   (V pre-transposed: row = d, contiguous keys)
        #pragma unroll
        for (int nt = 0; nt < 8; ++nt) {
            AF vf;
            const int drow = nt * 16 + l16;
            #pragma unroll
            for (int v = 0; v < 8; ++v) {
                int kk = (v >> 2) * 16 + hl * 8 + (v & 3) * 2;
                vf.u[v] = *(const unsigned int*)&vp[(size_t)drow * TT + jbase + kk];
            }
            O[nt].v = __builtin_amdgcn_wmma_f32_16x16x32_bf16(
                false, pf.v, false, vf.v, (short)0, O[nt].v, false, false);
        }
    }
    // normalize + store y [B][T][H*HD]
    #pragma unroll
    for (int nt = 0; nt < 8; ++nt)
        #pragma unroll
        for (int r = 0; r < 8; ++r) {
            int rowg = qbase + r + hl * 8;
            int col  = h * HD + nt * 16 + l16;
            y[((size_t)b * TT + rowg) * DIMC + col] = f2bf(O[nt].f[r] / lrow[r]);
        }
}

// out = x + bf2f(y)
__global__ void add_bf16_f32(const float* __restrict__ x,
                             const unsigned short* __restrict__ y,
                             float* __restrict__ out, int n)
{
    int i = blockIdx.x * blockDim.x + threadIdx.x;
    if (i < n) out[i] = x[i] + bf2f(y[i]);
}

// g = silu(g) * u   (bf16 in-place on g)
__global__ void silu_mul(unsigned short* __restrict__ g,
                         const unsigned short* __restrict__ u, int n)
{
    int i = blockIdx.x * blockDim.x + threadIdx.x;
    if (i >= n) return;
    float gv = bf2f(g[i]);
    float s  = gv * (1.f / (1.f + __expf(-gv)));
    g[i] = f2bf(s * bf2f(u[i]));
}

// ---------------------------------------------------------------------------
extern "C" void kernel_launch(void* const* d_in, const int* in_sizes, int n_in,
                              void* d_out, int out_size, void* d_ws, size_t ws_size,
                              hipStream_t stream)
{
    (void)in_sizes; (void)n_in; (void)out_size; (void)ws_size;
    const float* x   = (const float*)d_in[0];
    const float* rc  = (const float*)d_in[1];
    const float* rs  = (const float*)d_in[2];
    const float* wq  = (const float*)d_in[3];
    const float* wk  = (const float*)d_in[4];
    const float* wvw = (const float*)d_in[5];
    const float* wo  = (const float*)d_in[6];
    const float* wg  = (const float*)d_in[7];
    const float* wu  = (const float*)d_in[8];
    const float* wd  = (const float*)d_in[9];
    const float* anw = (const float*)d_in[10];
    const float* fnw = (const float*)d_in[11];
    float* out = (float*)d_out;

    char* ws = (char*)d_ws;
    const size_t MB = 1ull << 20;
    float*          x1   = (float*)         (ws + 0);        // 32 MB fp32 residual
    unsigned short* h16  = (unsigned short*)(ws + 32  * MB); // 16 MB
    unsigned short* wqT  = (unsigned short*)(ws + 48  * MB); //  8 MB each
    unsigned short* wkT  = (unsigned short*)(ws + 56  * MB);
    unsigned short* wvT  = (unsigned short*)(ws + 64  * MB);
    unsigned short* woT  = (unsigned short*)(ws + 72  * MB);
    unsigned short* wgT  = (unsigned short*)(ws + 48  * MB); // phase-B reuse, 32 MB
    unsigned short* wuT  = (unsigned short*)(ws + 80  * MB); // 32 MB
    unsigned short* wdT  = (unsigned short*)(ws + 112 * MB); // 32 MB
    unsigned short* q16  = (unsigned short*)(ws + 144 * MB); // 16 MB
    unsigned short* k16  = (unsigned short*)(ws + 160 * MB);
    unsigned short* v16  = (unsigned short*)(ws + 176 * MB);
    unsigned short* vT16 = (unsigned short*)(ws + 192 * MB);
    unsigned short* y16  = (unsigned short*)(ws + 208 * MB);
    unsigned short* o16  = (unsigned short*)(ws + 224 * MB);
    unsigned short* g16  = (unsigned short*)(ws + 144 * MB); // phase-B reuse, 64 MB
    unsigned short* u16  = (unsigned short*)(ws + 208 * MB); // 64 MB
    unsigned short* dn16 = (unsigned short*)(ws + 272 * MB); // 16 MB

    // ---- attention phase ----
    rmsnorm_cast<<<dim3(BT), 256, 0, stream>>>(x, anw, h16);
    cast_transpose<<<dim3(DIMC/32, DIMC/32), 256, 0, stream>>>(wq,  wqT, DIMC, DIMC);
    cast_transpose<<<dim3(DIMC/32, DIMC/32), 256, 0, stream>>>(wk,  wkT, DIMC, DIMC);
    cast_transpose<<<dim3(DIMC/32, DIMC/32), 256, 0, stream>>>(wvw, wvT, DIMC, DIMC);
    cast_transpose<<<dim3(DIMC/32, DIMC/32), 256, 0, stream>>>(wo,  woT, DIMC, DIMC);

    gemm_bf16<<<dim3(DIMC/128, BT/128), 256, 0, stream>>>(h16, wqT, q16, BT, DIMC, DIMC);
    gemm_bf16<<<dim3(DIMC/128, BT/128), 256, 0, stream>>>(h16, wkT, k16, BT, DIMC, DIMC);
    gemm_bf16<<<dim3(DIMC/128, BT/128), 256, 0, stream>>>(h16, wvT, v16, BT, DIMC, DIMC);

    const int ropeN = 2 * TT * NH * 64;
    rope_inplace<<<dim3(ropeN/256), 256, 0, stream>>>(q16, rc, rs);
    rope_inplace<<<dim3(ropeN/256), 256, 0, stream>>>(k16, rc, rs);
    transpose_v<<<dim3((2*TT*DIMC)/256), 256, 0, stream>>>(v16, vT16);

    attention<<<dim3(TT/64, 2*NH), 128, 0, stream>>>(q16, k16, vT16, y16);

    gemm_bf16<<<dim3(DIMC/128, BT/128), 256, 0, stream>>>(y16, woT, o16, BT, DIMC, DIMC);
    add_bf16_f32<<<dim3((BT*DIMC)/256), 256, 0, stream>>>(x, o16, x1, BT*DIMC);

    // ---- FFN phase ----
    rmsnorm_cast<<<dim3(BT), 256, 0, stream>>>(x1, fnw, h16);
    cast_transpose<<<dim3(HIDDENC/32, DIMC/32), 256, 0, stream>>>(wg, wgT, DIMC, HIDDENC);
    cast_transpose<<<dim3(HIDDENC/32, DIMC/32), 256, 0, stream>>>(wu, wuT, DIMC, HIDDENC);
    cast_transpose<<<dim3(DIMC/32, HIDDENC/32), 256, 0, stream>>>(wd, wdT, HIDDENC, DIMC);

    gemm_bf16<<<dim3(HIDDENC/128, BT/128), 256, 0, stream>>>(h16, wgT, g16, BT, HIDDENC, DIMC);
    gemm_bf16<<<dim3(HIDDENC/128, BT/128), 256, 0, stream>>>(h16, wuT, u16, BT, HIDDENC, DIMC);
    silu_mul<<<dim3((BT*HIDDENC)/256), 256, 0, stream>>>(g16, u16, BT*HIDDENC);

    gemm_bf16<<<dim3(DIMC/128, BT/128), 256, 0, stream>>>(g16, wdT, dn16, BT, DIMC, HIDDENC);
    add_bf16_f32<<<dim3((BT*DIMC)/256), 256, 0, stream>>>(x1, dn16, out, BT*DIMC);
}
